// GIN_3951369912896
// MI455X (gfx1250) — matrix operand order repaired and checked by reading, not compiled, not run on hardware
//
#include <hip/hip_runtime.h>

typedef __attribute__((ext_vector_type(16))) __bf16 v16bf;
typedef __attribute__((ext_vector_type(8)))  __bf16 v8bf;
typedef __attribute__((ext_vector_type(2)))  __bf16 v2bf;
typedef __attribute__((ext_vector_type(8)))  float  v8f;

#define N_NODES  100000
#define N_EDGES  1600000
#define DIM      64
#define LAYERS   5
#define NGRAPH   128
#define NOUT     10
#define BN_EPS   1e-5f
#define LDS_STR  72   // bf16 row stride: 144B, 16B-aligned, odd multiple of 16B -> bank spread
#define NTILES   (N_NODES / 16)   // 6250
#define WPB      8                // waves (16-row tiles) per block

// ---------------- utility ----------------
__global__ void zero_f32(float* __restrict__ p, int n) {
    int i = blockIdx.x * blockDim.x + threadIdx.x;
    if (i < n) p[i] = 0.f;
}

// one-shot: all layer weights -> bf16, TRANSPOSED (wt[n*64+k] = W[k*64+n])
__global__ void prep_weights(const float* __restrict__ W1all, const float* __restrict__ W2all,
                             __bf16* __restrict__ wt) {
    int i = blockIdx.x * blockDim.x + threadIdx.x;
    if (i >= LAYERS * 2 * 4096) return;
    int l = i >> 13;           // / 8192
    int rem = i & 8191;
    int m = rem >> 12;         // 0: W1, 1: W2
    int j = rem & 4095;
    int n = j >> 6, k = j & 63;
    const float* W = (m == 0 ? W1all : W2all) + (size_t)l * 4096;
    wt[(size_t)l * 8192 + m * 4096 + n * 64 + k] = (__bf16)W[k * 64 + n];
}

// ---------------- edge scatter: agg[dst] += x[src] ----------------
__global__ void scatter_add(const float* __restrict__ x,
                            const int* __restrict__ ei,
                            float* __restrict__ agg) {
    int t = blockIdx.x * blockDim.x + threadIdx.x;
    if (t >= N_EDGES * 16) return;
    int e = t >> 4;
    int c = (t & 15) * 4;
    int src = ei[e];
    int dst = ei[N_EDGES + e];
    const float4 v = *(const float4*)(x + (size_t)src * DIM + c);
    float* o = agg + (size_t)dst * DIM + c;
    atomicAdd(o + 0, v.x);
    atomicAdd(o + 1, v.y);
    atomicAdd(o + 2, v.z);
    atomicAdd(o + 3, v.w);
}

// load a 16-element bf16 fragment from two contiguous 16B LDS chunks
__device__ __forceinline__ v16bf frag2(const __bf16* p0, const __bf16* p1) {
    v8bf lo = *(const v8bf*)p0;
    v8bf hi = *(const v8bf*)p1;
    return __builtin_shufflevector(lo, hi, 0,1,2,3,4,5,6,7,8,9,10,11,12,13,14,15);
}

// ---------------- fused GIN MLP ----------------
// 256 threads = 8 waves; each wave owns one 16-node tile; weights shared in LDS.
__global__ void __launch_bounds__(256)
gin_mlp_wmma(const float* __restrict__ xin, const float* __restrict__ agg,
             const __bf16* __restrict__ wt1, const __bf16* __restrict__ wt2,
             const float* __restrict__ b1, const float* __restrict__ b2,
             float* __restrict__ h2out,
             float* __restrict__ bnsum, float* __restrict__ bnsq) {
    __shared__ __attribute__((aligned(16))) __bf16 sW1[64 * LDS_STR]; // transposed [n][k]
    __shared__ __attribute__((aligned(16))) __bf16 sW2[64 * LDS_STR];
    __shared__ __attribute__((aligned(16))) __bf16 sh [WPB][16 * LDS_STR];
    __shared__ __attribute__((aligned(16))) __bf16 sh1[WPB][16 * LDS_STR];

    const int tid  = threadIdx.x;
    const int wave = tid >> 5;
    const int lane = tid & 31;
    const int half = lane >> 4;
    const int row  = lane & 15;
    const int col  = lane & 15;

    // stage transposed bf16 weights via 16B chunks: 2 matrices * 64 rows * 8 chunks
    for (int c = tid; c < 2 * 64 * 8; c += 256) {
        int m  = c >> 9;
        int c2 = c & 511;
        int n  = c2 >> 3, kc = c2 & 7;
        const v8bf* src = (const v8bf*)((m ? wt2 : wt1) + n * 64 + kc * 8);
        *(v8bf*)((m ? sW2 : sW1) + n * LDS_STR + kc * 8) = *src;
    }

    const int tile   = blockIdx.x * WPB + wave;
    const bool active = tile < NTILES;
    const int m0     = tile * 16;

    if (active) {
        // stage h = x + agg as bf16 pairs (ds_store_b32)
        for (int i = lane; i < 16 * 32; i += 32) {
            int r = i >> 5, cp = (i & 31) * 2;
            size_t gi = (size_t)(m0 + r) * DIM + cp;
            v2bf p;
            p[0] = (__bf16)(xin[gi]     + agg[gi]);
            p[1] = (__bf16)(xin[gi + 1] + agg[gi + 1]);
            *(v2bf*)&sh[wave][r * LDS_STR + cp] = p;
        }
    }
    __syncthreads();

    const __bf16* shrow  = &sh[wave][row * LDS_STR];
    const __bf16* sh1row = &sh1[wave][row * LDS_STR];

    // ---- GEMM1: relu((x+agg) @ W1 + b1) -> sh1 ----
    if (active) {
        #pragma unroll
        for (int n0 = 0; n0 < 64; n0 += 16) {
            v8f acc = {};
            #pragma unroll
            for (int k0 = 0; k0 < 64; k0 += 32) {
                v16bf a = frag2(shrow + k0 + half * 8, shrow + k0 + 16 + half * 8);
                const __bf16* wcol = sW1 + (n0 + col) * LDS_STR + k0 + half * 8;
                v16bf b = frag2(wcol, wcol + 16);
                acc = __builtin_amdgcn_wmma_f32_16x16x32_bf16(
                        false, a, false, b, (short)0, acc, false, false);
            }
            #pragma unroll
            for (int r = 0; r < 8; ++r) {
                int rr = half * 8 + r, cc = n0 + col;
                float vv = acc[r] + b1[cc];
                vv = vv > 0.f ? vv : 0.f;
                sh1[wave][rr * LDS_STR + cc] = (__bf16)vv;
            }
        }
    }
    __syncthreads();

    // ---- GEMM2: sh1 @ W2 + b2 -> h2out (+ BN column partial sums) ----
    if (active) {
        #pragma unroll
        for (int n0 = 0; n0 < 64; n0 += 16) {
            v8f acc = {};
            #pragma unroll
            for (int k0 = 0; k0 < 64; k0 += 32) {
                v16bf a = frag2(sh1row + k0 + half * 8, sh1row + k0 + 16 + half * 8);
                const __bf16* wcol = sW2 + (n0 + col) * LDS_STR + k0 + half * 8;
                v16bf b = frag2(wcol, wcol + 16);
                acc = __builtin_amdgcn_wmma_f32_16x16x32_bf16(
                        false, a, false, b, (short)0, acc, false, false);
            }
            float csum = 0.f, csq = 0.f;
            #pragma unroll
            for (int r = 0; r < 8; ++r) {
                int rr = half * 8 + r, cc = n0 + col;
                float vv = acc[r] + b2[cc];
                h2out[(size_t)(m0 + rr) * DIM + cc] = vv;
                csum += vv;
                csq  += vv * vv;
            }
            csum += __shfl_xor(csum, 16, 32);
            csq  += __shfl_xor(csq,  16, 32);
            if (half == 0) {
                atomicAdd(&bnsum[n0 + col], csum);
                atomicAdd(&bnsq [n0 + col], csq);
            }
        }
    }
}

// ---------------- batchnorm (batch stats) + relu ----------------
__global__ void bn_relu(const float* __restrict__ h2,
                        const float* __restrict__ bnsum, const float* __restrict__ bnsq,
                        const float* __restrict__ gamma, const float* __restrict__ beta,
                        float* __restrict__ xout) {
    int i = blockIdx.x * blockDim.x + threadIdx.x;
    if (i >= N_NODES * DIM) return;
    int c = i & 63;
    const float invN = 1.f / (float)N_NODES;
    float mean = bnsum[c] * invN;
    float var  = bnsq[c] * invN - mean * mean;
    float v = gamma[c] * (h2[i] - mean) * rsqrtf(var + BN_EPS) + beta[c];
    xout[i] = v > 0.f ? v : 0.f;
}

// ---------------- per-graph mean pool ----------------
__global__ void pool_kernel(const float* __restrict__ x, const int* __restrict__ b,
                            float* __restrict__ psum, float* __restrict__ pcnt) {
    int i = blockIdx.x * blockDim.x + threadIdx.x;
    if (i >= N_NODES * DIM) return;
    int node = i >> 6, c = i & 63;
    int g = b[node];
    atomicAdd(&psum[g * 64 + c], x[i]);
    if (c == 0) atomicAdd(&pcnt[g], 1.f);
}

// ---------------- head MLP (tiny: 128x64 -> 64 -> 10) ----------------
__global__ void head_kernel(const float* __restrict__ psum, const float* __restrict__ pcnt,
                            const float* __restrict__ W1, const float* __restrict__ b1,
                            const float* __restrict__ W2, const float* __restrict__ b2,
                            float* __restrict__ out) {
    __shared__ float pooled[64];
    __shared__ float t[64];
    int g = blockIdx.x, j = threadIdx.x;
    float cnt = pcnt[g];
    cnt = cnt > 1.f ? cnt : 1.f;
    pooled[j] = psum[g * 64 + j] / cnt;
    __syncthreads();
    float acc = b1[j];
    for (int k = 0; k < 64; ++k) acc += pooled[k] * W1[k * 64 + j];
    t[j] = acc > 0.f ? acc : 0.f;
    __syncthreads();
    if (j < NOUT) {
        float acc2 = b2[j];
        for (int k = 0; k < 64; ++k) acc2 += t[k] * W2[k * NOUT + j];
        out[g * NOUT + j] = acc2;
    }
}

extern "C" void kernel_launch(void* const* d_in, const int* in_sizes, int n_in,
                              void* d_out, int out_size, void* d_ws, size_t ws_size,
                              hipStream_t stream) {
    const float* x     = (const float*)d_in[0];
    const int*   ei    = (const int*)  d_in[1];
    const int*   bidx  = (const int*)  d_in[2];
    const float* cW1   = (const float*)d_in[3];
    const float* cb1   = (const float*)d_in[4];
    const float* cW2   = (const float*)d_in[5];
    const float* cb2   = (const float*)d_in[6];
    const float* gamma = (const float*)d_in[7];
    const float* beta  = (const float*)d_in[8];
    const float* hW1   = (const float*)d_in[9];
    const float* hb1   = (const float*)d_in[10];
    const float* hW2   = (const float*)d_in[11];
    const float* hb2   = (const float*)d_in[12];
    float* out = (float*)d_out;

    const size_t NF = (size_t)N_NODES * DIM;
    float* ws    = (float*)d_ws;
    float* bufA  = ws;
    float* bufB  = bufA + NF;
    float* agg   = bufB + NF;
    float* h2    = agg  + NF;
    float* bnsum = h2   + NF;            // 64
    float* bnsq  = bnsum + 64;           // 64 (contiguous with bnsum)
    float* psum  = bnsq  + 64;           // G*64
    float* pcnt  = psum  + NGRAPH * 64;  // G (contiguous with psum)
    __bf16* wt   = (__bf16*)(pcnt + NGRAPH);  // L*2*4096 bf16 transposed weights

    prep_weights<<<(LAYERS * 2 * 4096 + 255) / 256, 256, 0, stream>>>(cW1, cW2, wt);

    const float* cur = x;
    float* nxt = bufA;
    for (int l = 0; l < LAYERS; ++l) {
        zero_f32<<<(int)((NF + 255) / 256), 256, 0, stream>>>(agg, (int)NF);
        zero_f32<<<1, 256, 0, stream>>>(bnsum, 128);
        scatter_add<<<(N_EDGES * 16 + 255) / 256, 256, 0, stream>>>(cur, ei, agg);
        gin_mlp_wmma<<<(NTILES + WPB - 1) / WPB, 256, 0, stream>>>(
            cur, agg,
            wt + (size_t)l * 8192, wt + (size_t)l * 8192 + 4096,
            cb1 + l * 64, cb2 + l * 64,
            h2, bnsum, bnsq);
        bn_relu<<<(int)((NF + 255) / 256), 256, 0, stream>>>(
            h2, bnsum, bnsq, gamma + l * 64, beta + l * 64, nxt);
        cur = nxt;
        nxt = (nxt == bufA) ? bufB : bufA;
    }

    zero_f32<<<(NGRAPH * 64 + NGRAPH + 255) / 256, 256, 0, stream>>>(psum, NGRAPH * 64 + NGRAPH);
    pool_kernel<<<(int)((NF + 255) / 256), 256, 0, stream>>>(cur, bidx, psum, pcnt);
    head_kernel<<<NGRAPH, 64, 0, stream>>>(psum, pcnt, hW1, hb1, hW2, hb2, out);
}
